// LoRAMultiheadAttention_85864986181639
// MI455X (gfx1250) — matrix operand order repaired and checked by reading, not compiled, not run on hardware
//
#include <hip/hip_runtime.h>
#include <hip/hip_bf16.h>

// ---------------------------------------------------------------------------
// LoRA Multihead Attention for gfx1250 (MI455X), bf16 WMMA pipeline.
//   T=2048, B=2, d=1024, H=16, hd=64, rank=16, scale = SCALE/RANK = 1.0
// ---------------------------------------------------------------------------

typedef __attribute__((ext_vector_type(16))) __bf16 v16bf;
typedef __attribute__((ext_vector_type(8)))  __bf16 bf16x8;
typedef __attribute__((ext_vector_type(8)))  float  v8f;

#define DEV __device__ __forceinline__

constexpr int T_LEN = 2048;
constexpr int BSZ   = 2;
constexpr int DM    = 1024;
constexpr int NH    = 16;
constexpr int HD    = 64;
constexpr int RNK   = 16;
constexpr int MROWS = T_LEN * BSZ;       // 4096
constexpr int N3D   = 3 * DM;            // 3072

union Frag { v16bf v; bf16x8 h[2]; };

DEV v8f wmma_bf16(v16bf a, v16bf b, v8f c) {
  return __builtin_amdgcn_wmma_f32_16x16x32_bf16(false, a, false, b,
                                                 (short)0, c, false, false);
}

// A operand (16x32, bf16, row-major source, ld in elements).
// Lane layout (ISA 7.12.2): lanes 0-15 -> row=lane, K={k0..k0+7, k0+16..k0+23}
//                           lanes 16-31 -> row=lane-16, K={k0+8.., k0+24..}
DEV v16bf load_a_frag(const __bf16* A, int ld, int row0, int k0) {
  int lane = threadIdx.x & 31;
  const __bf16* p = A + (size_t)(row0 + (lane & 15)) * ld + k0 + ((lane & 16) ? 8 : 0);
  Frag f;
  f.h[0] = *(const bf16x8*)(p);
  f.h[1] = *(const bf16x8*)(p + 16);
  return f.v;
}

// B operand (32x16, bf16). Source is column-major-B == row-major "Bt" where
// row `col` holds that output column's K values contiguously.
// Lanes 0-15: col=lane, K={k0..k0+7, k0+16..k0+23} ; lanes 16-31: col=lane-16, K={k0+8.., k0+24..}
DEV v16bf load_b_frag(const __bf16* Bt, int ld, int col0, int k0) {
  int lane = threadIdx.x & 31;
  const __bf16* p = Bt + (size_t)(col0 + (lane & 15)) * ld + k0 + ((lane & 16) ? 8 : 0);
  Frag f;
  f.h[0] = *(const bf16x8*)(p);
  f.h[1] = *(const bf16x8*)(p + 16);
  return f.v;
}

// ---------------------------------------------------------------------------
// Kernel 1: fold LoRA into QKV projection weights, convert to bf16.
// Wb[n][k] = W[n][k] + (k-band: sum_r k_a[r][k]*k_b[r][n-d])
//                    + (v-band: sum_r v_a[r][k]*v_b[r][n-2d])   (scale = 1.0)
// ---------------------------------------------------------------------------
__global__ __launch_bounds__(256) void fold_qkv_weights(
    const float* __restrict__ W, const float* __restrict__ ka,
    const float* __restrict__ kb, const float* __restrict__ va,
    const float* __restrict__ vb, __bf16* __restrict__ Wb) {
  int idx = blockIdx.x * 256 + threadIdx.x;      // over N3D*DM
  int n = idx >> 10;
  int k = idx & (DM - 1);
  float val = W[idx];
  if (n >= DM && n < 2 * DM) {
    int j = n - DM;
    float acc = 0.f;
#pragma unroll
    for (int r = 0; r < RNK; ++r) acc += ka[r * DM + k] * kb[r * DM + j];
    val += acc;                                   // SCALE/RANK == 1.0
  } else if (n >= 2 * DM) {
    int j = n - 2 * DM;
    float acc = 0.f;
#pragma unroll
    for (int r = 0; r < RNK; ++r) acc += va[r * DM + k] * vb[r * DM + j];
    val += acc;
  }
  Wb[idx] = (__bf16)val;
}

__global__ __launch_bounds__(256) void cvt_bf16_kernel(
    const float* __restrict__ src, __bf16* __restrict__ dst) {
  size_t i = (size_t)blockIdx.x * 256 + threadIdx.x;
  dst[i] = (__bf16)src[i];
}

// ---------------------------------------------------------------------------
// Kernel 2: QKV GEMM.  C[m][n] = sum_k Qb[m][k] * Wb[n][k] + bias[n]
// Epilogue scatters into attention-friendly layouts:
//   Qh[b][h][t][j] *= 1/sqrt(hd) ; Kh[b][h][t][j] ; Vt[b][h][j][t]
// Block: 256 thr = 8 waves (2 x 4); wave tile 32x32 (2x2 WMMA tiles).
// ---------------------------------------------------------------------------
__global__ __launch_bounds__(256) void gemm_qkv(
    const __bf16* __restrict__ A, const __bf16* __restrict__ Bt,
    const float* __restrict__ bias, __bf16* __restrict__ Qh,
    __bf16* __restrict__ Kh, __bf16* __restrict__ Vt) {
  int wave = threadIdx.x >> 5, lane = threadIdx.x & 31;
  int wm = wave >> 2, wn = wave & 3;
  int m0 = blockIdx.y * 64 + wm * 32;
  int n0 = blockIdx.x * 128 + wn * 32;

  v8f z = {};
  v8f acc00 = z, acc01 = z, acc10 = z, acc11 = z;

  for (int k = 0; k < DM; k += 32) {
    if (k + 32 < DM) {
      __builtin_prefetch(A + (size_t)(m0 + (lane & 15)) * DM + k + 32, 0, 0);
      __builtin_prefetch(Bt + (size_t)(n0 + (lane & 15)) * DM + k + 32, 0, 0);
    }
    v16bf a0 = load_a_frag(A, DM, m0, k);
    v16bf a1 = load_a_frag(A, DM, m0 + 16, k);
    v16bf b0 = load_b_frag(Bt, DM, n0, k);
    v16bf b1 = load_b_frag(Bt, DM, n0 + 16, k);
    acc00 = wmma_bf16(a0, b0, acc00);
    acc01 = wmma_bf16(a0, b1, acc01);
    acc10 = wmma_bf16(a1, b0, acc10);
    acc11 = wmma_bf16(a1, b1, acc11);
  }

  const v8f* accs[4] = {&acc00, &acc01, &acc10, &acc11};
#pragma unroll
  for (int mi = 0; mi < 2; ++mi)
#pragma unroll
    for (int ni = 0; ni < 2; ++ni) {
      const v8f& acc = *accs[mi * 2 + ni];
#pragma unroll
      for (int i = 0; i < 8; ++i) {
        int row = m0 + mi * 16 + i + ((lane >> 4) << 3);
        int col = n0 + ni * 16 + (lane & 15);
        float val = acc[i] + bias[col];
        int t = row >> 1, b = row & 1;                  // m = t*BSZ + b
        if (col < DM) {
          int h = col >> 6, j = col & 63;
          Qh[((size_t)((b << 4) + h) * T_LEN + t) * HD + j] = (__bf16)(val * 0.125f);
        } else if (col < 2 * DM) {
          int c = col - DM, h = c >> 6, j = c & 63;
          Kh[((size_t)((b << 4) + h) * T_LEN + t) * HD + j] = (__bf16)val;
        } else {
          int c = col - 2 * DM, h = c >> 6, j = c & 63;
          Vt[((size_t)((b << 4) + h) * HD + j) * T_LEN + t] = (__bf16)val;
        }
      }
    }
}

// ---------------------------------------------------------------------------
// Kernel 3: flash attention. grid=(T/128, B*H); 8 waves, 16 q-rows per wave.
// Streams keys/values in chunks of 32; online softmax; P staged via LDS to
// convert C-layout -> A-layout for the PV WMMAs.
// ---------------------------------------------------------------------------
__global__ __launch_bounds__(256) void attn_flash(
    const __bf16* __restrict__ Qh, const __bf16* __restrict__ Kh,
    const __bf16* __restrict__ Vt, __bf16* __restrict__ Ab) {
  __shared__ __align__(16) __bf16 p_lds[8][16][32];
  int wave = threadIdx.x >> 5, lane = threadIdx.x & 31;
  int bh = blockIdx.y;
  int b = bh >> 4, h = bh & 15;
  int t0 = blockIdx.x * 128 + wave * 16;

  const __bf16* Qbh = Qh + (size_t)bh * T_LEN * HD;
  const __bf16* Kbh = Kh + (size_t)bh * T_LEN * HD;
  const __bf16* Vbh = Vt + (size_t)bh * HD * T_LEN;

  v16bf qa0 = load_a_frag(Qbh, HD, t0, 0);
  v16bf qa1 = load_a_frag(Qbh, HD, t0, 32);

  v8f z = {};
  v8f o0 = z, o1 = z, o2 = z, o3 = z;
  float m[8], l[8];
#pragma unroll
  for (int i = 0; i < 8; ++i) { m[i] = -1e30f; l[i] = 0.f; }

  for (int j0 = 0; j0 < T_LEN; j0 += 32) {
    // S = Q (16x64) @ K^T (64x32)  -> two 16x16 tiles
    v8f s0 = z, s1 = z;
    s0 = wmma_bf16(qa0, load_b_frag(Kbh, HD, j0, 0), s0);
    s0 = wmma_bf16(qa1, load_b_frag(Kbh, HD, j0, 32), s0);
    s1 = wmma_bf16(qa0, load_b_frag(Kbh, HD, j0 + 16, 0), s1);
    s1 = wmma_bf16(qa1, load_b_frag(Kbh, HD, j0 + 16, 32), s1);

    float cf[8];
#pragma unroll
    for (int i = 0; i < 8; ++i) {
      float mx = fmaxf(s0[i], s1[i]);
#pragma unroll
      for (int d = 1; d < 16; d <<= 1) mx = fmaxf(mx, __shfl_xor(mx, d, 32));
      float mn = fmaxf(m[i], mx);
      cf[i] = __expf(m[i] - mn);
      float p0 = __expf(s0[i] - mn);
      float p1 = __expf(s1[i] - mn);
      float ls = p0 + p1;
#pragma unroll
      for (int d = 1; d < 16; d <<= 1) ls += __shfl_xor(ls, d, 32);
      l[i] = l[i] * cf[i] + ls;
      m[i] = mn;
      int row = i + ((lane >> 4) << 3);
      p_lds[wave][row][lane & 15]        = (__bf16)p0;
      p_lds[wave][row][16 + (lane & 15)] = (__bf16)p1;
    }
#pragma unroll
    for (int i = 0; i < 8; ++i) {
      o0[i] *= cf[i]; o1[i] *= cf[i]; o2[i] *= cf[i]; o3[i] *= cf[i];
    }
    asm volatile("s_wait_dscnt 0" ::: "memory");   // P store -> P fragment load
    v16bf pa = load_a_frag(&p_lds[wave][0][0], 32, 0, 0);
    o0 = wmma_bf16(pa, load_b_frag(Vbh, T_LEN, 0,  j0), o0);
    o1 = wmma_bf16(pa, load_b_frag(Vbh, T_LEN, 16, j0), o1);
    o2 = wmma_bf16(pa, load_b_frag(Vbh, T_LEN, 32, j0), o2);
    o3 = wmma_bf16(pa, load_b_frag(Vbh, T_LEN, 48, j0), o3);
  }

  v8f* outs[4] = {&o0, &o1, &o2, &o3};
#pragma unroll
  for (int t = 0; t < 4; ++t)
#pragma unroll
    for (int i = 0; i < 8; ++i) {
      int row = i + ((lane >> 4) << 3);
      int tq = t0 + row;
      int mrow = tq * BSZ + b;
      float val = (*outs[t])[i] / l[i];
      Ab[(size_t)mrow * DM + h * HD + t * 16 + (lane & 15)] = (__bf16)val;
    }
}

// ---------------------------------------------------------------------------
// Kernel 4: output projection. out[m][n] = sum_k Ab[m][k]*Wob[n][k] + ob[n]
// ---------------------------------------------------------------------------
__global__ __launch_bounds__(256) void gemm_out(
    const __bf16* __restrict__ A, const __bf16* __restrict__ Bt,
    const float* __restrict__ ob, float* __restrict__ out) {
  int wave = threadIdx.x >> 5, lane = threadIdx.x & 31;
  int wm = wave >> 2, wn = wave & 3;
  int m0 = blockIdx.y * 64 + wm * 32;
  int n0 = blockIdx.x * 128 + wn * 32;

  v8f z = {};
  v8f acc00 = z, acc01 = z, acc10 = z, acc11 = z;
  for (int k = 0; k < DM; k += 32) {
    if (k + 32 < DM) {
      __builtin_prefetch(A + (size_t)(m0 + (lane & 15)) * DM + k + 32, 0, 0);
      __builtin_prefetch(Bt + (size_t)(n0 + (lane & 15)) * DM + k + 32, 0, 0);
    }
    v16bf a0 = load_a_frag(A, DM, m0, k);
    v16bf a1 = load_a_frag(A, DM, m0 + 16, k);
    v16bf b0 = load_b_frag(Bt, DM, n0, k);
    v16bf b1 = load_b_frag(Bt, DM, n0 + 16, k);
    acc00 = wmma_bf16(a0, b0, acc00);
    acc01 = wmma_bf16(a0, b1, acc01);
    acc10 = wmma_bf16(a1, b0, acc10);
    acc11 = wmma_bf16(a1, b1, acc11);
  }
  const v8f* accs[4] = {&acc00, &acc01, &acc10, &acc11};
#pragma unroll
  for (int mi = 0; mi < 2; ++mi)
#pragma unroll
    for (int ni = 0; ni < 2; ++ni) {
      const v8f& acc = *accs[mi * 2 + ni];
#pragma unroll
      for (int i = 0; i < 8; ++i) {
        int row = m0 + mi * 16 + i + ((lane >> 4) << 3);
        int col = n0 + ni * 16 + (lane & 15);
        out[(size_t)row * DM + col] = acc[i] + ob[col];
      }
    }
}

// ---------------------------------------------------------------------------
extern "C" void kernel_launch(void* const* d_in, const int* in_sizes, int n_in,
                              void* d_out, int out_size, void* d_ws, size_t ws_size,
                              hipStream_t stream) {
  (void)in_sizes; (void)n_in; (void)out_size; (void)ws_size;
  const float* query = (const float*)d_in[0];
  const float* ipw   = (const float*)d_in[5];
  const float* ipb   = (const float*)d_in[6];
  const float* outw  = (const float*)d_in[7];
  const float* outb  = (const float*)d_in[8];
  const float* ka    = (const float*)d_in[9];
  const float* kb    = (const float*)d_in[10];
  const float* va    = (const float*)d_in[11];
  const float* vb    = (const float*)d_in[12];
  float* out = (float*)d_out;

  // Workspace carve-up (256B aligned regions).
  char* ws = (char*)d_ws;
  size_t off = 0;
  auto carve = [&](size_t bytes) {
    char* p = ws + off;
    off += (bytes + 255) & ~(size_t)255;
    return p;
  };
  __bf16* Wb  = (__bf16*)carve((size_t)N3D * DM * 2);          // folded qkv W
  __bf16* Qb  = (__bf16*)carve((size_t)MROWS * DM * 2);        // query bf16
  __bf16* Wob = (__bf16*)carve((size_t)DM * DM * 2);           // out_w bf16
  __bf16* Qh  = (__bf16*)carve((size_t)BSZ * NH * T_LEN * HD * 2);
  __bf16* Kh  = (__bf16*)carve((size_t)BSZ * NH * T_LEN * HD * 2);
  __bf16* Vt  = (__bf16*)carve((size_t)BSZ * NH * HD * T_LEN * 2);
  __bf16* Ab  = (__bf16*)carve((size_t)MROWS * DM * 2);        // attn out bf16

  // 1. fold LoRA into weights; convert activations/weights to bf16
  fold_qkv_weights<<<(N3D * DM) / 256, 256, 0, stream>>>(ipw, ka, kb, va, vb, Wb);
  cvt_bf16_kernel<<<(MROWS * DM) / 256, 256, 0, stream>>>(query, Qb);
  cvt_bf16_kernel<<<(DM * DM) / 256, 256, 0, stream>>>(outw, Wob);

  // 2. fused QKV projection (bias + head-split + q-scale + V-transpose)
  gemm_qkv<<<dim3(N3D / 128, MROWS / 64), 256, 0, stream>>>(Qb, Wb, ipb, Qh, Kh, Vt);

  // 3. flash attention
  attn_flash<<<dim3(T_LEN / 128, BSZ * NH), 256, 0, stream>>>(Qh, Kh, Vt, Ab);

  // 4. output projection -> f32
  gemm_out<<<dim3(DM / 128, MROWS / 64), 256, 0, stream>>>(Ab, Wob, outb, out);
}
